// ConvVQVAE_75754633167470
// MI455X (gfx1250) — compile-verified, matrix operand order checked by reading, not run on hardware
//
#include <hip/hip_runtime.h>
#include <hip/hip_bf16.h>

// ---------------------------------------------------------------------------
// VQ-VAE forward for MI455X (gfx1250, wave32, WMMA bf16 f32-accumulate).
// Bandwidth-bound (~44 GFLOP vs ~330 MB @ 23.3 TB/s): bf16 intermediates,
// WMMA for all GEMM stages, fragment-linear pre-packed weights (32B/lane
// loads), async global->LDS staging for the conv2 im2col tile.
// ---------------------------------------------------------------------------

typedef __attribute__((ext_vector_type(16))) __bf16 v16bf;
typedef __attribute__((ext_vector_type(8)))  __bf16 v8bf;
typedef __attribute__((ext_vector_type(8)))  float  v8f;

#define M_ROWS   200704          // 4096 * 7 * 7
#define N_TILES  3136            // M_ROWS / 64

// --- ISA 7.12.2 fragment K map: lanes 0-15 K = j&7 (+16 if j>=8); lanes 16-31 +8
__device__ __forceinline__ int kmap(int lane, int j) {
    return ((lane & 16) >> 1) + (j & 7) + ((j & 8) << 1);
}

// A fragment from row-major tile: per lane = two contiguous 16B loads + shuffle
__device__ __forceinline__ v16bf load_a16(const __bf16* p, int ld, int lane) {
    const __bf16* q = p + (lane & 15) * ld + ((lane & 16) >> 1);
    v8bf lo = *(const v8bf*)q;
    v8bf hi = *(const v8bf*)(q + 16);
    return __builtin_shufflevector(lo, hi, 0,1,2,3,4,5,6,7,8,9,10,11,12,13,14,15);
}

// B fragment from fragment-linear packed weights: one 32B load per lane
__device__ __forceinline__ v16bf load_b_packed(const __bf16* pk, int tile, int lane) {
    return *(const v16bf*)(pk + (((tile << 5) + lane) << 4));
}

__device__ __forceinline__ v8f wmma_bf16(v16bf a, v16bf b, v8f c) {
    return __builtin_amdgcn_wmma_f32_16x16x32_bf16(false, a, false, b, (short)0, c, false, false);
}

// --- gfx1250 async global->LDS (ISA 15.18.3 op 98, GVS mode) ---------------
__device__ __forceinline__ void async_ld_b128(unsigned ldsOff, unsigned gOff,
                                              const void* gBase) {
    asm volatile("global_load_async_to_lds_b128 %0, %1, %2"
                 :: "v"(ldsOff), "v"(gOff),
                    "s"((unsigned long long)(size_t)gBase)
                 : "memory");
}
__device__ __forceinline__ void wait_async0() {
    asm volatile("s_wait_asynccnt 0x0" ::: "memory");
}

// ---------------------------------------------------------------------------
// Kernel W: weight prep.  All B matrices converted to bf16 AND pre-swizzled
// into fragment-linear order [tile][lane][j] so GEMM kernels load v16bf
// directly.  Also: codebook bf16 (row-major), codebook norms, loss := 0.
// ---------------------------------------------------------------------------
__global__ __launch_bounds__(256) void prep_k(
    const float* __restrict__ w2, const float* __restrict__ w3,
    const float* __restrict__ cb, const float* __restrict__ dw0,
    const float* __restrict__ dw1,
    __bf16* __restrict__ w2pk, __bf16* __restrict__ w3pk,
    __bf16* __restrict__ cbbf, __bf16* __restrict__ dw0pk,
    __bf16* __restrict__ clspk, float* __restrict__ cnorm,
    float* __restrict__ loss)
{
    int t = blockIdx.x * 256 + threadIdx.x;          // 0..32767
    int lane = (t >> 4) & 31, j = t & 15, n16 = lane & 15;

    cbbf[t] = (__bf16)cb[t];                         // row-major codebook

    {   // w2 packed: 64 tiles = (kt 0..15) x (nt 0..3), K=512 N=64 row-major src
        int tile = t >> 9, kt = tile >> 2, nt = tile & 3;
        int k = kt * 32 + kmap(lane, j), n = nt * 16 + n16;
        w2pk[t] = (__bf16)w2[k * 64 + n];
    }
    if (t < 4096) {  // w3 / dw0 packed: 8 tiles = (kt 0..1) x (nt 0..3)
        int tile = t >> 9, kt = tile >> 2, nt = tile & 3;
        int k = kt * 32 + kmap(lane, j), n = nt * 16 + n16;
        w3pk[t]  = (__bf16)w3[k * 64 + n];
        dw0pk[t] = (__bf16)dw0[k * 64 + n];
    }
    {   // deconv1 class weights packed: 64 tiles = (cls 0..3)x(kt 0..7)x(nt 0..1)
        int tg = t >> 9, cls = tg >> 4, rest = tg & 15, kt = rest >> 1, nt = rest & 1;
        int py = cls >> 1, px = cls & 1;
        int k = kt * 32 + kmap(lane, j);
        int ci = k & 63, tt = k >> 6, ty = tt >> 1, tx = tt & 1;
        int ky = py + 2 * ty, kx = px + 2 * tx;
        int co = nt * 16 + n16;
        clspk[t] = (__bf16)dw1[((ky * 4 + kx) * 64 + ci) * 32 + co];
    }
    if (t < 512) {
        float s = 0.f;
        for (int d = 0; d < 64; ++d) { float v = cb[t * 64 + d]; s += v * v; }
        cnorm[t] = s;
    }
    if (t == 0) *loss = 0.f;
}

// ---------------------------------------------------------------------------
// Kernel A: conv1 (1->32, 4x4, s2, p1) + ReLU, direct f32 -> h1 bf16 NHWC.
// ---------------------------------------------------------------------------
__global__ __launch_bounds__(256) void conv1_k(
    const float* __restrict__ x, const float* __restrict__ w1,
    const float* __restrict__ b1, __bf16* __restrict__ h1)
{
    int t = blockIdx.x * 256 + threadIdx.x;          // ((b*14+oy)*14+ox)*32+co
    int co = t & 31, r = t >> 5;
    int ox = r % 14; r /= 14; int oy = r % 14; int b = r / 14;
    float acc = b1[co];
#pragma unroll
    for (int ky = 0; ky < 4; ++ky) {
        int iy = oy * 2 - 1 + ky;
        if (iy < 0 || iy >= 28) continue;
#pragma unroll
        for (int kx = 0; kx < 4; ++kx) {
            int ix = ox * 2 - 1 + kx;
            if (ix < 0 || ix >= 28) continue;
            acc += x[b * 784 + iy * 28 + ix] * w1[(ky * 4 + kx) * 32 + co];
        }
    }
    h1[t] = (__bf16)(acc > 0.f ? acc : 0.f);
}

// ---------------------------------------------------------------------------
// Kernel B: conv2 as im2col GEMM 200704x64x512 (bf16 WMMA) + bias + ReLU.
// 64-row M tile; in-bounds taps staged global->LDS with async b128 DMA,
// OOB taps zeroed with ds_store_b128.
// ---------------------------------------------------------------------------
__global__ __launch_bounds__(128) void conv2_k(
    const __bf16* __restrict__ h1, const __bf16* __restrict__ w2pk,
    const float* __restrict__ b2, __bf16* __restrict__ h2bf)
{
    __shared__ __bf16 sA[64 * 512];
    int tid = threadIdx.x, lane = tid & 31, wave = tid >> 5;
    int m0 = blockIdx.x * 64;

    {   // im2col fill: thread t -> row (t>>1), taps [(t&1)*8, +8)
        int r = tid >> 1, hf = tid & 1;
        int m = m0 + r;
        int b = m / 49, s = m % 49, oy = s / 7, ox = s % 7;
        unsigned sA0 = (unsigned)(size_t)(void*)&sA[0];
        v8bf z8 = {};
        for (int t4 = hf * 8; t4 < hf * 8 + 8; ++t4) {
            int ky = t4 >> 2, kx = t4 & 3;
            int iy = oy * 2 - 1 + ky, ix = ox * 2 - 1 + kx;
            unsigned ldsOff = sA0 + (unsigned)(r * 512 + t4 * 32) * 2u;
            if (iy >= 0 && iy < 14 && ix >= 0 && ix < 14) {
                unsigned gOff = (unsigned)(((b * 14 + iy) * 14 + ix) * 32) * 2u;
#pragma unroll
                for (int q = 0; q < 4; ++q)
                    async_ld_b128(ldsOff + q * 16u, gOff + q * 16u, h1);
            } else {
                v8bf* dst = (v8bf*)&sA[r * 512 + t4 * 32];
                dst[0] = z8; dst[1] = z8; dst[2] = z8; dst[3] = z8;
            }
        }
    }
    wait_async0();
    __syncthreads();

    v8f acc[4] = {};
    const __bf16* aRow = &sA[(wave * 16) * 512];
    for (int kt = 0; kt < 16; ++kt) {
        __builtin_prefetch(w2pk + ((kt + 1) << 11), 0, 3);   // next 4 packed tiles
        v16bf a = load_a16(aRow + kt * 32, 512, lane);
#pragma unroll
        for (int nt = 0; nt < 4; ++nt) {
            v16bf bb = load_b_packed(w2pk, kt * 4 + nt, lane);
            acc[nt] = wmma_bf16(a, bb, acc[nt]);
        }
    }
    int mw = m0 + wave * 16, hi = (lane & 16) >> 1;
#pragma unroll
    for (int nt = 0; nt < 4; ++nt) {
        int n = nt * 16 + (lane & 15);
        float bias = b2[n];
#pragma unroll
        for (int rr = 0; rr < 8; ++rr) {
            float v = acc[nt][rr] + bias;
            h2bf[(mw + rr + hi) * 64 + n] = (__bf16)(v > 0.f ? v : 0.f);
        }
    }
}

// ---------------------------------------------------------------------------
// Kernel C: conv3 (1x1, 64->64) GEMM + bias; writes z f32 (NCHW) + z bf16.
// ---------------------------------------------------------------------------
__global__ __launch_bounds__(128) void conv3_k(
    const __bf16* __restrict__ h2bf, const __bf16* __restrict__ w3pk,
    const float* __restrict__ b3, float* __restrict__ zout,
    __bf16* __restrict__ zbf)
{
    int lane = threadIdx.x & 31, wave = threadIdx.x >> 5;
    int m0 = blockIdx.x * 64 + wave * 16;
    v8f acc[4] = {};
    const __bf16* aBase = h2bf + m0 * 64;
#pragma unroll
    for (int kt = 0; kt < 2; ++kt) {
        v16bf a = load_a16(aBase + kt * 32, 64, lane);
#pragma unroll
        for (int nt = 0; nt < 4; ++nt) {
            v16bf bb = load_b_packed(w3pk, kt * 4 + nt, lane);
            acc[nt] = wmma_bf16(a, bb, acc[nt]);
        }
    }
    int hi = (lane & 16) >> 1;
#pragma unroll
    for (int nt = 0; nt < 4; ++nt) {
        int n = nt * 16 + (lane & 15);
        float bias = b3[n];
#pragma unroll
        for (int rr = 0; rr < 8; ++rr) {
            int m = m0 + rr + hi;
            float v = acc[nt][rr] + bias;
            int b = m / 49, s = m % 49;
            zout[(b * 64 + n) * 49 + s] = v;   // NCHW
            zbf[m * 64 + n] = (__bf16)v;
        }
    }
}

// ---------------------------------------------------------------------------
// Kernel D: VQ.  dot(z, c_n) via 64 WMMAs/wave (codebook rows ARE the
// K-vectors, so B fragments vector-load straight from row-major cbbf);
// argmin via per-lane running min + LDS cross-lane reduce (first-occurrence
// tie-break); writes indices, z_q (NCHW), accumulates 1.25*mean((zq-z)^2).
// ---------------------------------------------------------------------------
__global__ __launch_bounds__(128) void vq_k(
    const __bf16* __restrict__ zbf, const __bf16* __restrict__ cbbf,
    const float* __restrict__ cnorm, const float* __restrict__ cb,
    const float* __restrict__ zout, float* __restrict__ zqout,
    float* __restrict__ lossacc, float* __restrict__ indf,
    int* __restrict__ idxbuf)
{
    __shared__ float sSc[4][16][16];
    __shared__ int   sIx[4][16][16];
    int lane = threadIdx.x & 31, wave = threadIdx.x >> 5;
    int m0 = blockIdx.x * 64 + wave * 16;
    const __bf16* aBase = zbf + m0 * 64;
    v16bf a0 = load_a16(aBase, 64, lane);
    v16bf a1 = load_a16(aBase + 32, 64, lane);

    float best[8]; int bidx[8];
#pragma unroll
    for (int r = 0; r < 8; ++r) { best[r] = 3.4e38f; bidx[r] = 0; }

    for (int nt = 0; nt < 32; ++nt) {
        __builtin_prefetch(cbbf + (nt + 1) * 1024, 0, 3);
        v8f acc = {};
        v16bf b0 = load_a16(cbbf + (nt * 16) * 64, 64, lane);
        acc = wmma_bf16(a0, b0, acc);
        v16bf b1 = load_a16(cbbf + (nt * 16) * 64 + 32, 64, lane);
        acc = wmma_bf16(a1, b1, acc);
        int n = nt * 16 + (lane & 15);
        float cn = cnorm[n];
#pragma unroll
        for (int r = 0; r < 8; ++r) {
            float s = cn - 2.f * acc[r];                 // ||z||^2 constant per row
            if (s < best[r]) { best[r] = s; bidx[r] = n; }
        }
    }
    int hi = (lane & 16) >> 1;
#pragma unroll
    for (int r = 0; r < 8; ++r) {
        sSc[wave][r + hi][lane & 15] = best[r];
        sIx[wave][r + hi][lane & 15] = bidx[r];
    }
    __syncthreads();
    if (lane < 16) {
        int row = lane;
        float bs = sSc[wave][row][0]; int bi = sIx[wave][row][0];
        for (int c = 1; c < 16; ++c) {
            float s = sSc[wave][row][c]; int i2 = sIx[wave][row][c];
            if (s < bs || (s == bs && i2 < bi)) { bs = s; bi = i2; }
        }
        int m = m0 + row;
        idxbuf[m] = bi;
        indf[m] = (float)bi;
        int b = m / 49, s49 = m % 49;
        float part = 0.f;
        for (int d = 0; d < 64; ++d) {
            float q  = cb[bi * 64 + d];
            float zv = zout[(b * 64 + d) * 49 + s49];
            zqout[(b * 64 + d) * 49 + s49] = q;
            float df = q - zv; part += df * df;
        }
        atomicAdd(lossacc, part * (1.25f / (200704.f * 64.f)));
    }
}

// ---------------------------------------------------------------------------
// Kernel E: decoder 1x1 (64->64) GEMM; A rows gathered from codebook by index.
// ---------------------------------------------------------------------------
__global__ __launch_bounds__(128) void dec0_k(
    const int* __restrict__ idxbuf, const __bf16* __restrict__ cbbf,
    const __bf16* __restrict__ dw0pk, const float* __restrict__ db0,
    __bf16* __restrict__ d0bf)
{
    int lane = threadIdx.x & 31, wave = threadIdx.x >> 5;
    int m0 = blockIdx.x * 64 + wave * 16;
    int code = idxbuf[m0 + (lane & 15)];
    const __bf16* arow = cbbf + code * 64 + ((lane & 16) >> 1);
    v8f acc[4] = {};
#pragma unroll
    for (int kt = 0; kt < 2; ++kt) {
        v8bf lo = *(const v8bf*)(arow + kt * 32);
        v8bf hh = *(const v8bf*)(arow + kt * 32 + 16);
        v16bf a = __builtin_shufflevector(lo, hh, 0,1,2,3,4,5,6,7,8,9,10,11,12,13,14,15);
#pragma unroll
        for (int nt = 0; nt < 4; ++nt) {
            v16bf bb = load_b_packed(dw0pk, kt * 4 + nt, lane);
            acc[nt] = wmma_bf16(a, bb, acc[nt]);
        }
    }
    int hi = (lane & 16) >> 1;
#pragma unroll
    for (int nt = 0; nt < 4; ++nt) {
        int n = nt * 16 + (lane & 15);
        float bias = db0[n];
#pragma unroll
        for (int rr = 0; rr < 8; ++rr)
            d0bf[(m0 + rr + hi) * 64 + n] = (__bf16)(acc[nt][rr] + bias);
    }
}

// ---------------------------------------------------------------------------
// Kernel F: deconv1 (64->32, 4x4, s2 transposed) as 4 parity-class GEMMs,
// K = 4 taps x 64 ci; A-fragment = two conditional 16B loads per lane.
// ---------------------------------------------------------------------------
__global__ __launch_bounds__(128) void deconv1_k(
    const __bf16* __restrict__ d0bf, const __bf16* __restrict__ clspk,
    const float* __restrict__ db1, __bf16* __restrict__ d1)
{
    int lane = threadIdx.x & 31, wave = threadIdx.x >> 5;
    int cls = blockIdx.y, py = cls >> 1, px = cls & 1;
    int m0 = blockIdx.x * 64 + wave * 16;
    int mrow = m0 + (lane & 15);
    int b = mrow / 49, s = mrow % 49, oy2 = s / 7, ox2 = s % 7;

    v8f acc[2] = {};
    v8bf z8 = {};
    for (int kt = 0; kt < 8; ++kt) {
        int k0 = kt * 32 + ((lane & 16) >> 1);
        v8bf half[2];
#pragma unroll
        for (int h = 0; h < 2; ++h) {
            int k = k0 + h * 16;                       // run of 8 within one tap
            int ci = k & 63, tt = k >> 6, ty = tt >> 1, tx = tt & 1;
            int iy = oy2 + py - 1 + ty, ix = ox2 + px - 1 + tx;
            half[h] = z8;
            if (iy >= 0 && iy < 7 && ix >= 0 && ix < 7)
                half[h] = *(const v8bf*)(d0bf + ((b * 7 + iy) * 7 + ix) * 64 + ci);
        }
        v16bf a = __builtin_shufflevector(half[0], half[1],
                                          0,1,2,3,4,5,6,7,8,9,10,11,12,13,14,15);
#pragma unroll
        for (int nt = 0; nt < 2; ++nt) {
            v16bf bb = load_b_packed(clspk, cls * 16 + kt * 2 + nt, lane);
            acc[nt] = wmma_bf16(a, bb, acc[nt]);
        }
    }
    int hi = (lane & 16) >> 1;
#pragma unroll
    for (int nt = 0; nt < 2; ++nt) {
        int n = nt * 16 + (lane & 15);
        float bias = db1[n];
#pragma unroll
        for (int rr = 0; rr < 8; ++rr) {
            int m = m0 + rr + hi;
            int b2 = m / 49, ss = m % 49, oy2b = ss / 7, ox2b = ss % 7;
            int oy = 2 * oy2b + py, ox = 2 * ox2b + px;
            float v = acc[nt][rr] + bias;
            d1[((b2 * 14 + oy) * 14 + ox) * 32 + n] = (__bf16)(v > 0.f ? v : 0.f);
        }
    }
}

// ---------------------------------------------------------------------------
// Kernel G: deconv2 (32->1, 4x4, s2 transposed) + sigmoid -> recon (28x28).
// ---------------------------------------------------------------------------
__global__ __launch_bounds__(256) void deconv2_k(
    const __bf16* __restrict__ d1, const float* __restrict__ dw2,
    const float* __restrict__ db2, float* __restrict__ recon)
{
    int t = blockIdx.x * 256 + threadIdx.x;          // b*784 + oy*28 + ox
    int ox = t % 28, r = t / 28, oy = r % 28, b = r / 28;
    float acc = db2[0];
#pragma unroll
    for (int dty = 0; dty < 2; ++dty) {
        int ky = (oy & 1) + 2 * dty;
        int iy = (oy + ky - 2) >> 1;
        if (oy + ky < 2 || iy >= 14) continue;
#pragma unroll
        for (int dtx = 0; dtx < 2; ++dtx) {
            int kx = (ox & 1) + 2 * dtx;
            int ix = (ox + kx - 2) >> 1;
            if (ox + kx < 2 || ix >= 14) continue;
            const __bf16* src = &d1[((b * 14 + iy) * 14 + ix) * 32];
            const float*  w   = &dw2[(ky * 4 + kx) * 32];
#pragma unroll
            for (int ci = 0; ci < 32; ++ci) acc += (float)src[ci] * w[ci];
        }
    }
    recon[t] = 1.f / (1.f + expf(-acc));
}

// ---------------------------------------------------------------------------
extern "C" void kernel_launch(void* const* d_in, const int* in_sizes, int n_in,
                              void* d_out, int out_size, void* d_ws, size_t ws_size,
                              hipStream_t stream) {
    (void)in_sizes; (void)n_in; (void)out_size; (void)ws_size;
    const float* x    = (const float*)d_in[0];
    const float* w1   = (const float*)d_in[1];
    const float* b1   = (const float*)d_in[2];
    const float* w2   = (const float*)d_in[3];
    const float* b2   = (const float*)d_in[4];
    const float* w3   = (const float*)d_in[5];
    const float* b3   = (const float*)d_in[6];
    const float* cb   = (const float*)d_in[7];
    const float* dw0  = (const float*)d_in[8];
    const float* db0  = (const float*)d_in[9];
    const float* dw1  = (const float*)d_in[10];
    const float* db1  = (const float*)d_in[11];
    const float* dw2  = (const float*)d_in[12];
    const float* db2  = (const float*)d_in[13];

    // d_out layout (floats): recon | z | z_q | vq_loss | indices
    float* out    = (float*)d_out;
    float* recon  = out;                               // 3,211,264
    float* zout   = out + 3211264;                     // 12,845,056
    float* zqout  = out + 16056320;                    // 12,845,056
    float* loss   = out + 28901376;                    // 1
    float* indf   = out + 28901377;                    // 200,704

    // workspace (~104 MB), with buffer reuse (d1 aliases h1, d0 aliases h2)
    char* ws = (char*)d_ws;
    __bf16* h1    = (__bf16*)(ws);                     // 51,380,224 B
    __bf16* d1    = h1;                                // reuse after conv2
    __bf16* h2bf  = (__bf16*)(ws + 51380224u);         // 25,690,112 B
    __bf16* d0bf  = h2bf;                              // reuse after conv3
    __bf16* zbf   = (__bf16*)(ws + 77070336u);         // 25,690,112 B
    char*   wsw   = ws + 102760448u;
    __bf16* w2pk  = (__bf16*)(wsw);                    // 65,536 B (packed)
    __bf16* w3pk  = (__bf16*)(wsw + 65536u);           //  8,192 B (packed)
    __bf16* cbbf  = (__bf16*)(wsw + 73728u);           // 65,536 B (row-major)
    __bf16* dw0pk = (__bf16*)(wsw + 139264u);          //  8,192 B (packed)
    __bf16* clspk = (__bf16*)(wsw + 147456u);          // 65,536 B (packed)
    float*  cnorm = (float*)(wsw + 212992u);           //  2,048 B
    int*    idxb  = (int*)(wsw + 215040u);             // 802,816 B

    prep_k   <<<128, 256, 0, stream>>>(w2, w3, cb, dw0, dw1,
                                       w2pk, w3pk, cbbf, dw0pk, clspk, cnorm, loss);
    conv1_k  <<<100352, 256, 0, stream>>>(x, w1, b1, h1);
    conv2_k  <<<N_TILES, 128, 0, stream>>>(h1, w2pk, b2, h2bf);
    conv3_k  <<<N_TILES, 128, 0, stream>>>(h2bf, w3pk, b3, zout, zbf);
    vq_k     <<<N_TILES, 128, 0, stream>>>(zbf, cbbf, cnorm, cb, zout,
                                           zqout, loss, indf, idxb);
    dec0_k   <<<N_TILES, 128, 0, stream>>>(idxb, cbbf, dw0pk, db0, d0bf);
    deconv1_k<<<dim3(N_TILES, 4), 128, 0, stream>>>(d0bf, clspk, db1, d1);
    deconv2_k<<<12544, 256, 0, stream>>>(d1, dw2, db2, recon);
}
